// ContentAwareMambaFilter_73847667687892
// MI455X (gfx1250) — matrix operand order repaired and checked by reading, not compiled, not run on hardware
//
#include <hip/hip_runtime.h>
#include <hip/hip_bf16.h>
#include <math.h>

// ---------------- problem constants ----------------
#define DIM      512
#define D_STATE  16
#define D_CONV   4
#define D_INNER  1024
#define DT_RANK  32
#define BATCH    4
#define SEQ      2048
#define MTOK     (BATCH * SEQ)        // 8192 tokens

typedef __bf16 bf16_t;
typedef __attribute__((ext_vector_type(16))) __bf16 v16bf;
typedef __attribute__((ext_vector_type(8)))  float  v8f;

// ---------------- small helpers ----------------
__device__ __forceinline__ float dev_sigmoid(float x) { return 1.0f / (1.0f + __expf(-x)); }
__device__ __forceinline__ float dev_silu(float x)    { return x * dev_sigmoid(x); }
__device__ __forceinline__ float dev_softplus(float x){ return (x > 20.0f) ? x : log1pf(__expf(x)); }

// ---------------- 1) saliency gate ----------------
__global__ void gate_kernel(const float* __restrict__ x, const float* __restrict__ sal,
                            const float* __restrict__ W1, const float* __restrict__ b1,
                            const float* __restrict__ W2, const float* __restrict__ b2,
                            bf16_t* __restrict__ xg)
{
    __shared__ float red[128];
    const int tok = blockIdx.x;
    const int j   = threadIdx.x;
    const float s = sal[tok];
    const float h = fmaxf(0.0f, W1[j] * s + b1[j]);
    red[j] = h * W2[j];
    __syncthreads();
    for (int k = 64; k > 0; k >>= 1) {
        if (j < k) red[j] += red[j + k];
        __syncthreads();
    }
    const float gate  = dev_sigmoid(red[0] + b2[0]);
    const float scale = 1.0f - 0.2f * gate;
    const size_t base = (size_t)tok * DIM;
    for (int c = j; c < DIM; c += 128)
        xg[base + c] = (bf16_t)(x[base + c] * scale);
}

// ---------------- 2) f32 -> bf16 convert ----------------
__global__ void cvt_bf16_kernel(const float* __restrict__ in, bf16_t* __restrict__ out, int n)
{
    for (int i = blockIdx.x * blockDim.x + threadIdx.x; i < n; i += gridDim.x * blockDim.x)
        out[i] = (bf16_t)in[i];
}

// ---------------- 3) WMMA GEMM: C(M,N) = A(M,K) * W(N,K)^T, bf16 in / f32 acc ----------------
// Register-blocked: each wave computes a 32x64 C region (2 M-tiles x 4 N-tiles),
// 8 v_wmma per 32-wide K step, reusing 2 A fragments across 4 B fragments.
// Requires: M % 32 == 0, N % 64 == 0, K % 32 == 0 (true for all call sites).
// act: 0 = none, 1 = softplus(x + bias[n])
__global__ void gemm_bf16_wmma(const bf16_t* __restrict__ A, const bf16_t* __restrict__ W,
                               float* __restrict__ Cf, bf16_t* __restrict__ Cb,
                               const float* __restrict__ bias,
                               int M, int N, int K, int act)
{
    const int w       = (blockIdx.x * blockDim.x + threadIdx.x) >> 5;   // global wave id
    const int lane    = threadIdx.x & 31;
    const int ngroups = N >> 6;                                         // N / 64
    const int nwaves  = (M >> 5) * ngroups;
    if (w >= nwaves) return;

    const int m2 = w / ngroups;
    const int ng = w % ngroups;
    const int M0 = m2 << 5;      // 32-row region
    const int N0 = ng << 6;      // 64-col region

    const int half = lane >> 4;  // 0: lanes 0-15, 1: lanes 16-31
    const int r    = lane & 15;

    // A fragments: rows M0+r (tile 0) and M0+16+r (tile 1)
    // lane-half K sub-pattern: {0..15} / {16..31}
    const bf16_t* Arow0 = A + (size_t)(M0 + r) * K + half * 16;
    const bf16_t* Arow1 = Arow0 + (size_t)16 * K;
    // B fragments: column N0 + j*16 + r of W; half selects 16 contiguous K values
    const bf16_t* Wrow0 = W + (size_t)(N0 + r) * K + half * 16;

    v8f c[2][4] = {};

    for (int k = 0; k < K; k += 32) {
        v16bf a0, a1, b0, b1, b2, b3;
        {
            const uint4* p = reinterpret_cast<const uint4*>(Arow0 + k);
            ((uint4*)&a0)[0] = p[0];  ((uint4*)&a0)[1] = p[1];
        }
        {
            const uint4* p = reinterpret_cast<const uint4*>(Arow1 + k);
            ((uint4*)&a1)[0] = p[0];  ((uint4*)&a1)[1] = p[1];
        }
        {
            const uint4* p = reinterpret_cast<const uint4*>(Wrow0 + k);
            ((uint4*)&b0)[0] = p[0];  ((uint4*)&b0)[1] = p[1];
        }
        {
            const uint4* p = reinterpret_cast<const uint4*>(Wrow0 + (size_t)16 * K + k);
            ((uint4*)&b1)[0] = p[0];  ((uint4*)&b1)[1] = p[1];
        }
        {
            const uint4* p = reinterpret_cast<const uint4*>(Wrow0 + (size_t)32 * K + k);
            ((uint4*)&b2)[0] = p[0];  ((uint4*)&b2)[1] = p[1];
        }
        {
            const uint4* p = reinterpret_cast<const uint4*>(Wrow0 + (size_t)48 * K + k);
            ((uint4*)&b3)[0] = p[0];  ((uint4*)&b3)[1] = p[1];
        }
        __builtin_prefetch(Arow0 + k + 128, 0, 1);   // global_prefetch_b8
        __builtin_prefetch(Wrow0 + k + 128, 0, 1);

        c[0][0] = __builtin_amdgcn_wmma_f32_16x16x32_bf16(false, a0, false, b0, (short)0, c[0][0], false, false);
        c[0][1] = __builtin_amdgcn_wmma_f32_16x16x32_bf16(false, a0, false, b1, (short)0, c[0][1], false, false);
        c[0][2] = __builtin_amdgcn_wmma_f32_16x16x32_bf16(false, a0, false, b2, (short)0, c[0][2], false, false);
        c[0][3] = __builtin_amdgcn_wmma_f32_16x16x32_bf16(false, a0, false, b3, (short)0, c[0][3], false, false);
        c[1][0] = __builtin_amdgcn_wmma_f32_16x16x32_bf16(false, a1, false, b0, (short)0, c[1][0], false, false);
        c[1][1] = __builtin_amdgcn_wmma_f32_16x16x32_bf16(false, a1, false, b1, (short)0, c[1][1], false, false);
        c[1][2] = __builtin_amdgcn_wmma_f32_16x16x32_bf16(false, a1, false, b2, (short)0, c[1][2], false, false);
        c[1][3] = __builtin_amdgcn_wmma_f32_16x16x32_bf16(false, a1, false, b3, (short)0, c[1][3], false, false);
    }

    // Epilogue. C layout: VGPR v -> row (subtile M base) + half*8 + v, column (subtile N base) + r
#pragma unroll
    for (int i = 0; i < 2; ++i) {
#pragma unroll
        for (int j = 0; j < 4; ++j) {
            const int col = N0 + j * 16 + r;
            const float bv = (act == 1) ? bias[col] : 0.0f;
#pragma unroll
            for (int v = 0; v < 8; ++v) {
                const int row = M0 + i * 16 + half * 8 + v;
                float val = c[i][j][v];
                if (act == 1) val = dev_softplus(val + bv);
                const size_t offc = (size_t)row * N + col;
                if (Cf) Cf[offc] = val;
                if (Cb) Cb[offc] = (bf16_t)val;
            }
        }
    }
}

// ---------------- 4) depthwise causal conv (k=4) + SiLU ----------------
__global__ void conv_silu_kernel(const bf16_t* __restrict__ xz, const float* __restrict__ cw,
                                 const float* __restrict__ cb,
                                 float* __restrict__ xcf, bf16_t* __restrict__ xcb)
{
    const int n = MTOK * D_INNER;
    for (int i = blockIdx.x * blockDim.x + threadIdx.x; i < n; i += gridDim.x * blockDim.x) {
        const int d   = i % D_INNER;
        const int row = i / D_INNER;           // b*SEQ + t
        const int t   = row % SEQ;
        const int b   = row / SEQ;
        float acc = cb[d];
#pragma unroll
        for (int k = 0; k < D_CONV; ++k) {
            const int tt = t - (D_CONV - 1) + k;
            if (tt >= 0)
                acc += cw[d * D_CONV + k] *
                       (float)xz[((size_t)(b * SEQ + tt)) * (2 * D_INNER) + d];
        }
        const float v = dev_silu(acc);
        xcf[i] = v;
        xcb[i] = (bf16_t)v;
    }
}

// ---------------- 5) extract dbc[..., :DT_RANK] as bf16 ----------------
__global__ void dtslice_kernel(const float* __restrict__ dbc, bf16_t* __restrict__ out)
{
    const int n = MTOK * DT_RANK;
    for (int i = blockIdx.x * blockDim.x + threadIdx.x; i < n; i += gridDim.x * blockDim.x) {
        const int row = i / DT_RANK;
        const int c   = i % DT_RANK;
        out[i] = (bf16_t)dbc[(size_t)row * (DT_RANK + 2 * D_STATE) + c];
    }
}

// ---------------- 6) sequential selective scan: one thread per (b, d) ----------------
__global__ void scan_kernel(const float* __restrict__ dt, const float* __restrict__ xcf,
                            const float* __restrict__ dbc, const float* __restrict__ A_log,
                            float* __restrict__ ys)
{
    const int tid = blockIdx.x * blockDim.x + threadIdx.x;
    if (tid >= BATCH * D_INNER) return;
    const int b = tid / D_INNER;
    const int d = tid % D_INNER;

    float An[D_STATE];
#pragma unroll
    for (int nn = 0; nn < D_STATE; ++nn) An[nn] = -__expf(A_log[d * D_STATE + nn]);

    float h[D_STATE];
#pragma unroll
    for (int nn = 0; nn < D_STATE; ++nn) h[nn] = 0.0f;

    for (int t = 0; t < SEQ; ++t) {
        const size_t row = (size_t)b * SEQ + t;
        const float dtv = dt[row * D_INNER + d];
        const float xcv = xcf[row * D_INNER + d];
        const float* bc = dbc + row * (DT_RANK + 2 * D_STATE);
        float y = 0.0f;
#pragma unroll
        for (int nn = 0; nn < D_STATE; ++nn) {
            const float dA = __expf(dtv * An[nn]);
            const float hn = dA * h[nn] + dtv * bc[DT_RANK + nn] * xcv;
            h[nn] = hn;
            y += hn * bc[DT_RANK + D_STATE + nn];
        }
        ys[row * D_INNER + d] = y;
    }
}

// ---------------- 7) y = (ys + D*xc) * silu(z)  -> bf16 ----------------
__global__ void ymerge_kernel(const float* __restrict__ ys, const float* __restrict__ xcf,
                              const bf16_t* __restrict__ xz, const float* __restrict__ Dp,
                              bf16_t* __restrict__ yb)
{
    const int n = MTOK * D_INNER;
    for (int i = blockIdx.x * blockDim.x + threadIdx.x; i < n; i += gridDim.x * blockDim.x) {
        const int d   = i % D_INNER;
        const int row = i / D_INNER;
        const float z = (float)xz[(size_t)row * (2 * D_INNER) + D_INNER + d];
        const float y = (ys[i] + Dp[d] * xcf[i]) * dev_silu(z);
        yb[i] = (bf16_t)y;
    }
}

// ---------------- 8) residual + LayerNorm ----------------
__global__ void ln_kernel(const float* __restrict__ x, const float* __restrict__ xf,
                          const float* __restrict__ gamma, const float* __restrict__ beta,
                          float* __restrict__ out)
{
    __shared__ float ssum[256];
    __shared__ float ssq[256];
    const int tok = blockIdx.x;
    const int j   = threadIdx.x;
    const size_t base = (size_t)tok * DIM;

    const float r0 = x[base + j]       + 0.3f * xf[base + j];
    const float r1 = x[base + j + 256] + 0.3f * xf[base + j + 256];
    ssum[j] = r0 + r1;
    ssq[j]  = r0 * r0 + r1 * r1;
    __syncthreads();
    for (int k = 128; k > 0; k >>= 1) {
        if (j < k) { ssum[j] += ssum[j + k]; ssq[j] += ssq[j + k]; }
        __syncthreads();
    }
    const float mean = ssum[0] * (1.0f / DIM);
    const float var  = ssq[0] * (1.0f / DIM) - mean * mean;
    const float inv  = rsqrtf(var + 1e-5f);
    out[base + j]       = (r0 - mean) * inv * gamma[j]       + beta[j];
    out[base + j + 256] = (r1 - mean) * inv * gamma[j + 256] + beta[j + 256];
}

// ---------------- launch ----------------
static inline size_t align256(size_t x) { return (x + 255) & ~(size_t)255; }
static inline int gemm_blocks(int M, int N) {
    const int waves = (M / 32) * (N / 64);
    return (waves + 7) / 8;            // 8 waves (256 threads) per block
}

extern "C" void kernel_launch(void* const* d_in, const int* in_sizes, int n_in,
                              void* d_out, int out_size, void* d_ws, size_t ws_size,
                              hipStream_t stream)
{
    const float* x      = (const float*)d_in[0];
    const float* sal    = (const float*)d_in[1];
    const float* W1     = (const float*)d_in[2];
    const float* b1     = (const float*)d_in[3];
    const float* W2     = (const float*)d_in[4];
    const float* b2     = (const float*)d_in[5];
    const float* W_in   = (const float*)d_in[6];
    const float* conv_w = (const float*)d_in[7];
    const float* conv_b = (const float*)d_in[8];
    const float* W_x    = (const float*)d_in[9];
    const float* W_dt   = (const float*)d_in[10];
    const float* b_dt   = (const float*)d_in[11];
    const float* A_log  = (const float*)d_in[12];
    const float* Dp     = (const float*)d_in[13];
    const float* W_out  = (const float*)d_in[14];
    const float* gamma  = (const float*)d_in[15];
    const float* beta   = (const float*)d_in[16];
    float* out = (float*)d_out;

    // workspace layout (bytes)
    char* ws = (char*)d_ws;
    size_t off = 0;
    auto alloc = [&](size_t bytes) { char* p = ws + off; off = align256(off + bytes); return p; };

    bf16_t* xg_b    = (bf16_t*)alloc((size_t)MTOK * DIM * 2);                 // x_gated bf16
    bf16_t* Win_b   = (bf16_t*)alloc((size_t)2 * D_INNER * DIM * 2);          // W_in bf16
    bf16_t* Wx_b    = (bf16_t*)alloc((size_t)(DT_RANK + 2 * D_STATE) * D_INNER * 2);
    bf16_t* Wdt_b   = (bf16_t*)alloc((size_t)D_INNER * DT_RANK * 2);
    bf16_t* Wout_b  = (bf16_t*)alloc((size_t)DIM * D_INNER * 2);
    bf16_t* xz_b    = (bf16_t*)alloc((size_t)MTOK * 2 * D_INNER * 2);         // xz bf16
    float*  xc_f    = (float*)alloc((size_t)MTOK * D_INNER * 4);              // conv+silu f32
    bf16_t* xc_b    = (bf16_t*)alloc((size_t)MTOK * D_INNER * 2);             // conv+silu bf16
    float*  dbc_f   = (float*)alloc((size_t)MTOK * (DT_RANK + 2 * D_STATE) * 4);
    bf16_t* dt_in_b = (bf16_t*)alloc((size_t)MTOK * DT_RANK * 2);
    float*  dt_f    = (float*)alloc((size_t)MTOK * D_INNER * 4);              // softplus dt
    float*  ys_f    = (float*)alloc((size_t)MTOK * D_INNER * 4);              // scan output
    bf16_t* y_b     = (bf16_t*)alloc((size_t)MTOK * D_INNER * 2);             // gated y bf16
    float*  xf_f    = (float*)alloc((size_t)MTOK * DIM * 4);                  // mamba output
    (void)ws_size; (void)n_in; (void)in_sizes; (void)out_size;

    // 1) saliency gate -> x_gated (bf16)
    gate_kernel<<<MTOK, 128, 0, stream>>>(x, sal, W1, b1, W2, b2, xg_b);

    // 2) weight conversions
    cvt_bf16_kernel<<<512, 256, 0, stream>>>(W_in,  Win_b,  2 * D_INNER * DIM);
    cvt_bf16_kernel<<<64,  256, 0, stream>>>(W_x,   Wx_b,   (DT_RANK + 2 * D_STATE) * D_INNER);
    cvt_bf16_kernel<<<32,  256, 0, stream>>>(W_dt,  Wdt_b,  D_INNER * DT_RANK);
    cvt_bf16_kernel<<<256, 256, 0, stream>>>(W_out, Wout_b, DIM * D_INNER);

    // 3) xz = x_gated @ W_in^T  (M=8192, N=2048, K=512) -> bf16
    gemm_bf16_wmma<<<gemm_blocks(MTOK, 2 * D_INNER), 256, 0, stream>>>(
        xg_b, Win_b, nullptr, xz_b, nullptr, MTOK, 2 * D_INNER, DIM, 0);

    // 4) depthwise conv + SiLU on xi
    conv_silu_kernel<<<4096, 256, 0, stream>>>(xz_b, conv_w, conv_b, xc_f, xc_b);

    // 5) dbc = xc @ W_x^T  (M=8192, N=64, K=1024) -> f32
    gemm_bf16_wmma<<<gemm_blocks(MTOK, DT_RANK + 2 * D_STATE), 256, 0, stream>>>(
        xc_b, Wx_b, dbc_f, nullptr, nullptr, MTOK, DT_RANK + 2 * D_STATE, D_INNER, 0);

    // 6) slice dbc[:, :32] -> bf16
    dtslice_kernel<<<1024, 256, 0, stream>>>(dbc_f, dt_in_b);

    // 7) dt = softplus(dt_in @ W_dt^T + b_dt)  (M=8192, N=1024, K=32) -> f32, fused epilogue
    gemm_bf16_wmma<<<gemm_blocks(MTOK, D_INNER), 256, 0, stream>>>(
        dt_in_b, Wdt_b, dt_f, nullptr, b_dt, MTOK, D_INNER, DT_RANK, 1);

    // 8) selective scan: one thread per (b, d)
    scan_kernel<<<(BATCH * D_INNER + 255) / 256, 256, 0, stream>>>(dt_f, xc_f, dbc_f, A_log, ys_f);

    // 9) y = (ys + D*xc) * silu(z) -> bf16
    ymerge_kernel<<<4096, 256, 0, stream>>>(ys_f, xc_f, xz_b, Dp, y_b);

    // 10) x_filtered = y @ W_out^T  (M=8192, N=512, K=1024) -> f32
    gemm_bf16_wmma<<<gemm_blocks(MTOK, DIM), 256, 0, stream>>>(
        y_b, Wout_b, xf_f, nullptr, nullptr, MTOK, DIM, D_INNER, 0);

    // 11) residual + LayerNorm
    ln_kernel<<<MTOK, 256, 0, stream>>>(x, xf_f, gamma, beta, out);
}
